// FullBCModel_82841329205930
// MI455X (gfx1250) — matrix operand order repaired and checked by reading, not compiled, not run on hardware
//
#include <hip/hip_runtime.h>
#include <hip/hip_bf16.h>
#include <math.h>

typedef __bf16 v16bf __attribute__((ext_vector_type(16)));
typedef float  v8f   __attribute__((ext_vector_type(8)));

#define FREQf 64.0f
#define NBC   14
#define NACL  10
#define NACG  35
#define T_IN  32768
#define K0    20
#define K1    32
#define KAC   52
#define TLOOP (T_IN + K1 - 1)   // 32799 scan steps

// ---- workspace float-offset map ----
enum : int {
  OFF_B = 0, BCN_B = 16, SLOPE_B = 32, P01_B = 48, P12_B = 64,
  IPCAP_B = 80, INVRRP_B = 96, IP0_B = 112, RRP0_B = 128, SCAL = 144,
  TAPS = 160,              // 16 x 20
  ACLK = 480,              // 16 x 64 (zero padded)
  ACGK = 1504,             // 48 x 64 (zero padded)
  W2ACL = 4576,            // 16a x 16b  (bc_acl_w)
  W2ACG = 4832,            // 48g x 16b  (bc_acg_w)
  WP1 = 5600,              // 16b x 16a  (-exp acl_bc_w)
  WPG = 5856,              // 16b x 48g  (-exp acg_bc_w)
  WLG = 6624,              // 48g x 16l  (-exp acl_acg_w)
  KGLU = 7392,             // 32
  ACLOFF = 7424, ACLSLOPE = 7440,
  ACGOFF = 7456, ACGSLOPE = 7504,
  RELBASE = 8192           // rel[2][TLOOP][16]
};

__device__ __forceinline__ float sigf(float x, float off, float sl) {
  return 1.0f / (1.0f + __expf(-sl * (x - off)));
}

// ------------------------------------------------------------------
// Kernel 1: parameter preprocessing into workspace tables
// ------------------------------------------------------------------
__global__ __launch_bounds__(256)
void setup_kernel(const float* x_unused,
                  const float* sbias, const float* sscale,
                  const float* lspeed, const float* sig_off, const float* lslope,
                  const float* lp01, const float* lp12, const float* lipcap,
                  const float* lrrpcap, const float* ipst, const float* rrpst,
                  const float* laclbc, const float* lbcacl,
                  const float* lacltr, const float* lacltd,
                  const float* laclsl, const float* acloff,
                  const float* lacgbc, const float* lbcacg,
                  const float* lacgtr, const float* lacgtd,
                  const float* lacgsl, const float* acgoff,
                  const float* laclacg, const float* bcnoff,
                  float* ws)
{
  const int tid = threadIdx.x;

  if (tid < 16) {                      // per-BC scalar tables (+ acl sigmoid params)
    int b = tid; bool v = b < NBC;
    float so = v ? sig_off[b] : 0.f;
    ws[OFF_B + b]  = so;
    ws[BCN_B + b]  = v ? (so + bcnoff[b]) : 0.f;
    ws[SLOPE_B + b] = v ? __expf(lslope[b]) : 0.f;
    ws[P01_B + b]  = v ? __expf(lp01[b]) : 0.f;
    ws[P12_B + b]  = v ? __expf(lp12[b]) : 0.f;
    float ic = v ? __expf(lipcap[b]) : 1.f;
    float rc = v ? __expf(lrrpcap[b]) : 1.f;
    ws[IPCAP_B + b]  = ic;
    ws[INVRRP_B + b] = 1.f / rc;
    ws[IP0_B + b]  = v ? ic / (1.f + __expf(-ipst[b]))  : 0.f;
    ws[RRP0_B + b] = v ? rc / (1.f + __expf(-rrpst[b])) : 0.f;
    bool va = b < NACL;
    ws[ACLOFF + b]   = va ? acloff[b] : 0.f;
    ws[ACLSLOPE + b] = va ? __expf(laclsl[b]) : 0.f;
  }
  if (tid >= 16 && tid < 32) {         // biphasic BC taps (normalized, x cell_type)
    int b = tid - 16;
    if (b < NBC) {
      float sp = __expf(lspeed[b]);
      float ct = (b < 5) ? -1.f : 1.f;
      float row[K0]; float nrm = 0.f;
      for (int j = 0; j < K0; ++j) {
        float ts = ((float)j / FREQf) * sp;
        float a1 = ts / 0.05f, a2 = ts / 0.1f;
        float k = a1 * a1 * __expf(-a1) - 0.8f * a2 * a2 * __expf(-a2);
        k *= ct; row[j] = k; nrm += k * k;
      }
      float inv = rsqrtf(nrm);
      for (int j = 0; j < K0; ++j) ws[TAPS + b * K0 + j] = row[j] * inv;
    } else {
      for (int j = 0; j < K0; ++j) ws[TAPS + b * K0 + j] = 0.f;
    }
  }
  if (tid >= 32 && tid < 48) {         // ACL double-exp kernels, L2-normalized
    int a = tid - 32;
    if (a < NACL) {
      float tr = __expf(lacltr[a]), td = __expf(lacltd[a]);
      float c = (td + tr) / (td * tr);
      float row[KAC]; float nrm = 0.f;
      for (int k = 0; k < KAC; ++k) {
        float tt = 0.8f - (float)k / FREQf;
        float kv = __expf(-tt / td) - __expf(-c * tt);
        row[k] = kv; nrm += kv * kv;
      }
      float inv = rsqrtf(nrm);
      for (int k = 0; k < 64; ++k)
        ws[ACLK + a * 64 + k] = (k < KAC) ? row[k] * inv : 0.f;
    } else {
      for (int k = 0; k < 64; ++k) ws[ACLK + a * 64 + k] = 0.f;
    }
  }
  if (tid >= 48 && tid < 96) {         // ACG kernels + acg sigmoid params
    int g = tid - 48;
    if (g < NACG) {
      float tr = __expf(lacgtr[g]), td = __expf(lacgtd[g]);
      float c = (td + tr) / (td * tr);
      float row[KAC]; float nrm = 0.f;
      for (int k = 0; k < KAC; ++k) {
        float tt = 0.8f - (float)k / FREQf;
        float kv = __expf(-tt / td) - __expf(-c * tt);
        row[k] = kv; nrm += kv * kv;
      }
      float inv = rsqrtf(nrm);
      for (int k = 0; k < 64; ++k)
        ws[ACGK + g * 64 + k] = (k < KAC) ? row[k] * inv : 0.f;
    } else {
      for (int k = 0; k < 64; ++k) ws[ACGK + g * 64 + k] = 0.f;
    }
    ws[ACGOFF + g]   = (g < NACG) ? acgoff[g] : 0.f;
    ws[ACGSLOPE + g] = (g < NACG) ? __expf(lacgsl[g]) : 0.f;
  }
  if (tid == 96) {                     // stimulus affine + iGluSnFR kernel
    ws[SCAL]     = __expf(sscale[0]);
    ws[SCAL + 1] = sbias[0];
    float row[K1]; float s = 0.f;
    for (int j = 0; j < K1; ++j) {
      float tg = (float)j / FREQf;
      float kv = (1.f - __expf(-tg / 0.02f)) * __expf(-tg / 0.1f);
      row[j] = kv; s += kv;
    }
    for (int j = 0; j < K1; ++j) ws[KGLU + j] = row[j] / s;
  }
  // elementwise weight tables (zero padded)
  for (int idx = tid; idx < 256; idx += 256) {      // W2ACL [a][b]
    int a = idx >> 4, b = idx & 15;
    ws[W2ACL + idx] = (a < NACL && b < NBC) ? __expf(lbcacl[a * NBC + b]) : 0.f;
  }
  for (int idx = tid; idx < 768; idx += 256) {      // W2ACG [g][b]
    int g = idx >> 4, b = idx & 15;
    ws[W2ACG + idx] = (g < NACG && b < NBC) ? __expf(lbcacg[g * NBC + b]) : 0.f;
  }
  for (int idx = tid; idx < 256; idx += 256) {      // WP1 [b][a]
    int b = idx >> 4, a = idx & 15;
    ws[WP1 + idx] = (b < NBC && a < NACL) ? -__expf(laclbc[b * NACL + a]) : 0.f;
  }
  for (int idx = tid; idx < 768; idx += 256) {      // WPG [b][g]
    int b = idx / 48, g = idx % 48;
    ws[WPG + idx] = (b < NBC && g < NACG) ? -__expf(lacgbc[b * NACG + g]) : 0.f;
  }
  for (int idx = tid; idx < 768; idx += 256) {      // WLG [g][l]
    int g = idx >> 4, l = idx & 15;
    ws[WLG + idx] = (g < NACG && l < NACL) ? -__expf(laclacg[g * NACL + l]) : 0.f;
  }
}

// ------------------------------------------------------------------
// Kernel 2: single-wave sequential scan, WMMA feedback convolutions
// ------------------------------------------------------------------
__global__ __launch_bounds__(32)
void scan_kernel(const float* __restrict__ xg, float* __restrict__ ws)
{
  __shared__ float  ring[2][16][64];   // release history rings (b rows padded to 16)
  __shared__ v16bf  Bfrag[8][32];      // constant kernel B-fragments (WMMA layout)
  __shared__ float  aclL[2][16];
  __shared__ float  acgL[48];

  const int lane = threadIdx.x;
  const int n    = lane & 15;          // column / cell index within tile
  const int half = lane >> 4;

  for (int i = lane; i < 2 * 16 * 64; i += 32) ((float*)ring)[i] = 0.f;

  // Build constant B-fragments: B[k][col] with col=n, K=half*16+e (+chunk*32)
  #pragma unroll
  for (int f = 0; f < 8; ++f) {
    const float* kb; int chunk;
    if (f < 2) { kb = ws + ACLK + n * 64; chunk = f; }
    else       { int tl = (f - 2) >> 1; chunk = (f - 2) & 1;
                 kb = ws + ACGK + (tl * 16 + n) * 64; }
    v16bf bv;
    #pragma unroll
    for (int e = 0; e < 16; ++e)
      bv[e] = (__bf16)kb[chunk * 32 + half * 16 + e];
    Bfrag[f][lane] = bv;
  }

  // per-lane parameters
  float offB = ws[OFF_B + n], bcnB = ws[BCN_B + n], slB = ws[SLOPE_B + n];
  float p01 = ws[P01_B + n], p12 = ws[P12_B + n];
  float ipcap = ws[IPCAP_B + n], invrrp = ws[INVRRP_B + n];
  float ip1 = ws[IP0_B + n], rrp1 = ws[RRP0_B + n];
  float ip2 = ip1, rrp2 = rrp1;
  float Asc = ws[SCAL], Bof = ws[SCAL + 1];
  float tap[K0], tapsum = 0.f;
  #pragma unroll
  for (int j = 0; j < K0; ++j) { tap[j] = ws[TAPS + n * K0 + j]; tapsum += tap[j]; }
  float w2acl[8], w2acg[3][8];
  #pragma unroll
  for (int r = 0; r < 8; ++r) w2acl[r] = ws[W2ACL + n * 16 + r + 8 * half];
  #pragma unroll
  for (int tl = 0; tl < 3; ++tl)
    #pragma unroll
    for (int r = 0; r < 8; ++r)
      w2acg[tl][r] = ws[W2ACG + (tl * 16 + n) * 16 + r + 8 * half];
  float wp1v[NACL];
  #pragma unroll
  for (int a = 0; a < NACL; ++a) wp1v[a] = ws[WP1 + n * 16 + a];
  float wpgv[NACG];
  #pragma unroll
  for (int g = 0; g < NACG; ++g) wpgv[g] = ws[WPG + n * 48 + g];
  float wlgv[3][NACL];
  #pragma unroll
  for (int tl = 0; tl < 3; ++tl)
    #pragma unroll
    for (int l = 0; l < NACL; ++l) wlgv[tl][l] = ws[WLG + (tl * 16 + n) * 16 + l];
  float aclOf = ws[ACLOFF + n], aclSl = ws[ACLSLOPE + n];
  float acgOf[3], acgSl[3];
  #pragma unroll
  for (int tl = 0; tl < 3; ++tl) { acgOf[tl] = ws[ACGOFF + tl * 16 + n];
                                   acgSl[tl] = ws[ACGSLOPE + tl * 16 + n]; }
  float* relg = ws + RELBASE;

  __syncthreads();

  for (int t = 0; t < TLOOP; ++t) {
    __builtin_prefetch(xg + ((t + 64 < T_IN) ? t + 64 : T_IN - 1), 0, 0);

    // stimulus FIR drive (wave-uniform x reads; edge padding via clamp)
    float dr = 0.f;
    #pragma unroll
    for (int j = 0; j < K0; ++j) {
      int i0 = t - 31 - j; i0 = i0 < 0 ? 0 : i0;
      dr += tap[j] * xg[i0];
    }
    float xi = Asc * dr + Bof * tapsum;

    // A-fragments: buf[b][k] from ring, contiguous k = 16*half + e (matches Bfrag)
    v16bf a0, a1, b0, b1;
    {
      int base = t + 12;                         // (t-52) mod 64
      #pragma unroll
      for (int e = 0; e < 16; ++e) {
        int kl = 16 * half + e;
        a0[e] = (__bf16)ring[0][n][(base + kl) & 63];
        a1[e] = (__bf16)ring[0][n][(base + 32 + kl) & 63];
        b0[e] = (__bf16)ring[1][n][(base + kl) & 63];
        b1[e] = (__bf16)ring[1][n][(base + 32 + kl) & 63];
      }
    }
    v8f z = {};
    v8f d1 = __builtin_amdgcn_wmma_f32_16x16x32_bf16(false, a0, false, Bfrag[0][lane], (short)0, z,  false, false);
    d1     = __builtin_amdgcn_wmma_f32_16x16x32_bf16(false, a1, false, Bfrag[1][lane], (short)0, d1, false, false);
    v8f d2 = __builtin_amdgcn_wmma_f32_16x16x32_bf16(false, b0, false, Bfrag[0][lane], (short)0, z,  false, false);
    d2     = __builtin_amdgcn_wmma_f32_16x16x32_bf16(false, b1, false, Bfrag[1][lane], (short)0, d2, false, false);
    v8f g0 = __builtin_amdgcn_wmma_f32_16x16x32_bf16(false, b0, false, Bfrag[2][lane], (short)0, z,  false, false);
    g0     = __builtin_amdgcn_wmma_f32_16x16x32_bf16(false, b1, false, Bfrag[3][lane], (short)0, g0, false, false);
    v8f g1 = __builtin_amdgcn_wmma_f32_16x16x32_bf16(false, b0, false, Bfrag[4][lane], (short)0, z,  false, false);
    g1     = __builtin_amdgcn_wmma_f32_16x16x32_bf16(false, b1, false, Bfrag[5][lane], (short)0, g1, false, false);
    v8f g2 = __builtin_amdgcn_wmma_f32_16x16x32_bf16(false, b0, false, Bfrag[6][lane], (short)0, z,  false, false);
    g2     = __builtin_amdgcn_wmma_f32_16x16x32_bf16(false, b1, false, Bfrag[7][lane], (short)0, g2, false, false);

    // stage 2: acl_in[a] = sum_b w[a,b]*G[b,a]  (D: m=r+8*half, n=col)
    float s1 = 0.f, s2 = 0.f;
    #pragma unroll
    for (int r = 0; r < 8; ++r) { s1 += w2acl[r] * d1[r]; s2 += w2acl[r] * d2[r]; }
    s1 += __shfl_xor(s1, 16, 32);
    s2 += __shfl_xor(s2, 16, 32);
    float o1 = sigf(s1, aclOf, aclSl);
    float o2 = sigf(s2, aclOf, aclSl);
    if (half == 0 && n < NACL) { aclL[0][n] = o1; aclL[1][n] = o2; }
    __syncthreads();

    // ACG: kernel term + ACL->ACG coupling, sigmoid
    float sg0, sg1, sg2;
    {
      float s;
      s = 0.f;
      #pragma unroll
      for (int r = 0; r < 8; ++r) s += w2acg[0][r] * g0[r];
      s += __shfl_xor(s, 16, 32);
      #pragma unroll
      for (int l = 0; l < NACL; ++l) s += wlgv[0][l] * aclL[1][l];
      sg0 = sigf(s, acgOf[0], acgSl[0]);
      s = 0.f;
      #pragma unroll
      for (int r = 0; r < 8; ++r) s += w2acg[1][r] * g1[r];
      s += __shfl_xor(s, 16, 32);
      #pragma unroll
      for (int l = 0; l < NACL; ++l) s += wlgv[1][l] * aclL[1][l];
      sg1 = sigf(s, acgOf[1], acgSl[1]);
      s = 0.f;
      #pragma unroll
      for (int r = 0; r < 8; ++r) s += w2acg[2][r] * g2[r];
      s += __shfl_xor(s, 16, 32);
      #pragma unroll
      for (int l = 0; l < NACL; ++l) s += wlgv[2][l] * aclL[1][l];
      sg2 = sigf(s, acgOf[2], acgSl[2]);
    }
    if (half == 0) { acgL[n] = sg0; acgL[16 + n] = sg1; acgL[32 + n] = sg2; }
    __syncthreads();

    // BC drives, two-pool ribbon release, ring + global staging
    float acc1 = xi, acc2 = xi;
    #pragma unroll
    for (int a = 0; a < NACL; ++a) {
      acc1 += wp1v[a] * aclL[0][a];
      acc2 += wp1v[a] * aclL[1][a];
    }
    #pragma unroll
    for (int g = 0; g < NACG; ++g) acc2 += wpgv[g] * acgL[g];
    float p1 = sigf(acc1, offB, slB);
    float p2 = sigf(acc2, bcnB, slB);
    float rel1 = p1 * rrp1;
    { float t12 = p12 * ip1 * (1.f - rrp1 * invrrp);
      float t01 = p01 * (ipcap - ip1);
      ip1 += t01 - t12; rrp1 += t12 - rel1; }
    float rel2 = p2 * rrp2;
    { float t12 = p12 * ip2 * (1.f - rrp2 * invrrp);
      float t01 = p01 * (ipcap - ip2);
      ip2 += t01 - t12; rrp2 += t12 - rel2; }
    if (half == 0 && n < NBC) {
      ring[0][n][t & 63] = rel1;
      ring[1][n][t & 63] = rel2;
      relg[(0 * TLOOP + t) * 16 + n] = rel1;
      relg[(1 * TLOOP + t) * 16 + n] = rel2;
    }
    __syncthreads();
  }
}

// ------------------------------------------------------------------
// Kernel 3: iGluSnFR readout convolution (valid, K1=32)
// ------------------------------------------------------------------
__global__ __launch_bounds__(256)
void conv_kernel(const float* __restrict__ ws, float* __restrict__ out)
{
  int gid = blockIdx.x * blockDim.x + threadIdx.x;
  if (gid >= 2 * NBC * T_IN) return;
  int i  = gid & (T_IN - 1);
  int pc = gid >> 15;          // p*NBC + c
  int c  = pc % NBC;
  int p  = pc / NBC;
  const float* kg  = ws + KGLU;
  const float* rel = ws + RELBASE + p * TLOOP * 16;
  float s = 0.f;
  #pragma unroll
  for (int j = 0; j < K1; ++j) s += kg[j] * rel[(i + 31 - j) * 16 + c];
  out[gid] = s;
}

// ------------------------------------------------------------------
extern "C" void kernel_launch(void* const* d_in, const int* in_sizes, int n_in,
                              void* d_out, int out_size, void* d_ws, size_t ws_size,
                              hipStream_t stream) {
  const float* x        = (const float*)d_in[0];
  const float* sbias    = (const float*)d_in[1];
  const float* sscale   = (const float*)d_in[2];
  const float* lspeed   = (const float*)d_in[3];
  const float* sigoff   = (const float*)d_in[4];
  const float* lslope   = (const float*)d_in[5];
  const float* lp01     = (const float*)d_in[6];
  const float* lp12     = (const float*)d_in[7];
  const float* lipcap   = (const float*)d_in[8];
  const float* lrrpcap  = (const float*)d_in[9];
  const float* ipst     = (const float*)d_in[10];
  const float* rrpst    = (const float*)d_in[11];
  const float* laclbc   = (const float*)d_in[12];
  const float* lbcacl   = (const float*)d_in[13];
  const float* lacltr   = (const float*)d_in[14];
  const float* lacltd   = (const float*)d_in[15];
  const float* laclsl   = (const float*)d_in[16];
  const float* acloff   = (const float*)d_in[17];
  const float* lacgbc   = (const float*)d_in[18];
  const float* lbcacg   = (const float*)d_in[19];
  const float* lacgtr   = (const float*)d_in[20];
  const float* lacgtd   = (const float*)d_in[21];
  const float* lacgsl   = (const float*)d_in[22];
  const float* acgoff   = (const float*)d_in[23];
  const float* laclacg  = (const float*)d_in[24];
  const float* bcnoff   = (const float*)d_in[25];
  float* ws = (float*)d_ws;

  setup_kernel<<<1, 256, 0, stream>>>(x, sbias, sscale, lspeed, sigoff, lslope,
                                      lp01, lp12, lipcap, lrrpcap, ipst, rrpst,
                                      laclbc, lbcacl, lacltr, lacltd, laclsl, acloff,
                                      lacgbc, lbcacg, lacgtr, lacgtd, lacgsl, acgoff,
                                      laclacg, bcnoff, ws);
  scan_kernel<<<1, 32, 0, stream>>>(x, ws);
  int total = 2 * NBC * T_IN;
  conv_kernel<<<(total + 255) / 256, 256, 0, stream>>>(ws, (float*)d_out);
}
